// multi_head_attention_42631845380626
// MI455X (gfx1250) — matrix operand order repaired and checked by reading, not compiled, not run on hardware
//
#include <hip/hip_runtime.h>
#include <hip/hip_bf16.h>
#include <math.h>

// ---------------------------------------------------------------------------
// Relative-position MHA for MI455X (gfx1250, wave32, WMMA).
// All GEMMs run on v_wmma_f32_16x16x32_f16 (f16 in, f32 accumulate).
// Edge terms are restructured as GEMMs batched over the query index i.
// Each wave owns a 16x64 output tile (1 A-frag -> 4 WMMAs), and the k-loops
// are software-pipelined (register double buffering) so WMMAs never sit
// behind an s_wait_loadcnt 0.
// ---------------------------------------------------------------------------

typedef __attribute__((ext_vector_type(16))) _Float16 v16h;
typedef __attribute__((ext_vector_type(8)))  _Float16 v8h;
typedef __attribute__((ext_vector_type(8)))  float    v8f;
typedef __attribute__((ext_vector_type(4)))  float    v4f;

#define D_MODEL 1024
#define N_HEAD  16
#define DK      64
#define B_      8
#define S_      512
#define BH      (B_ * N_HEAD)   // 128

// ---- WMMA fragment helpers (wave32 layouts per CDNA5 ISA 7.12.2) ----------

// A (16x32 f16), row-major source, ld = lda halves.
// lane<16: row=lane, K = {0..7, 16..23}; lane>=16: row=lane-16, K={8..15,24..31}
static __device__ __forceinline__ v16h load_a_f16(const _Float16* base, int lda,
                                                  int mbase, int k0, int lid) {
    const _Float16* p = base + (size_t)(mbase + (lid & 15)) * lda + k0 + ((lid >> 4) << 3);
    v16h a;
    *(v8h*)&a       = *(const v8h*)p;        // K   .. K+7
    *((v8h*)&a + 1) = *(const v8h*)(p + 16); // K+16.. K+23
    return a;
}

// Raw f32 A fragment (load now, convert later -> pipelinable).
struct a16f { v4f x0, x1, x2, x3; };
static __device__ __forceinline__ a16f load_a_f32_raw(const float* base, int lda,
                                                      int mbase, int k0, int lid) {
    const float* p = base + (size_t)(mbase + (lid & 15)) * lda + k0 + ((lid >> 4) << 3);
    a16f r;
    r.x0 = *(const v4f*)p;
    r.x1 = *(const v4f*)(p + 4);
    r.x2 = *(const v4f*)(p + 16);
    r.x3 = *(const v4f*)(p + 20);
    return r;
}
static __device__ __forceinline__ v16h cvt_a(const a16f& r) {
    v16h a;
#pragma unroll
    for (int t = 0; t < 4; ++t) {
        a[t]      = (_Float16)r.x0[t];
        a[t + 4]  = (_Float16)r.x1[t];
        a[t + 8]  = (_Float16)r.x2[t];
        a[t + 12] = (_Float16)r.x3[t];
    }
    return a;
}

// B (32x16 f16) from a transposed ("BT[n][k]") layout; lane half h holds
// column n, K = {8h..8h+7, 16+8h..16+8h+7} (matches A) as two 16-byte loads.
static __device__ __forceinline__ v16h load_bt_f16(const _Float16* base, int ldb,
                                                   int nbase, int k0, int lid) {
    const _Float16* p = base + (size_t)(nbase + (lid & 15)) * ldb + k0 + ((lid >> 4) << 3);
    v16h b;
    *(v8h*)&b       = *(const v8h*)p;
    *((v8h*)&b + 1) = *(const v8h*)(p + 16);
    return b;
}

static __device__ __forceinline__ v8f wmma_f16(v16h a, v16h b, v8f c) {
    return __builtin_amdgcn_wmma_f32_16x16x32_f16(false, a, false, b, (short)0, c,
                                                  false, false);
}

// ---- small prep kernels ---------------------------------------------------

// dst[b][c][r] = (f16) src[b][r][c]   (LDS-tiled transpose + convert)
__global__ void k_transpose_cvt(const float* __restrict__ src, _Float16* __restrict__ dst,
                                int rows, int cols) {
    __shared__ float t[16][17];
    const float* s = src + (size_t)blockIdx.z * rows * cols;
    _Float16*    d = dst + (size_t)blockIdx.z * rows * cols;
    int r0 = blockIdx.y * 16, c0 = blockIdx.x * 16;
    int tx = threadIdx.x, ty = threadIdx.y;
    t[ty][tx] = s[(size_t)(r0 + ty) * cols + c0 + tx];
    __syncthreads();
    d[(size_t)(c0 + ty) * rows + r0 + tx] = (_Float16)t[tx][ty];
}

// elementwise f32 -> f16 (layout preserved)
__global__ void k_cvt_f16(const float* __restrict__ src, _Float16* __restrict__ dst,
                          long n) {
    long i = (long)blockIdx.x * blockDim.x + threadIdx.x;
    long stride = (long)gridDim.x * blockDim.x;
    for (; i < n; i += stride) dst[i] = (_Float16)src[i];
}

// ---- QKV projection: X[4096,1024] @ W -> Qh/Kh [bh][s][64], Vt [bh][64][s] -
// scale 1/sqrt(64) is folded into Q so it applies to both q.kT and q.e terms.
// Each wave: 16(m) x 64(n) output tile, K=1024, double-buffered k-loop.
__global__ void __launch_bounds__(128) k_qkv(
    const float* __restrict__ X,
    const _Float16* __restrict__ WqT, const _Float16* __restrict__ WkT,
    const _Float16* __restrict__ WvT,
    const float* __restrict__ bq, const float* __restrict__ bk,
    const float* __restrict__ bv,
    _Float16* __restrict__ Qh, _Float16* __restrict__ Kh, _Float16* __restrict__ Vt) {
    int lid = threadIdx.x & 31, wid = threadIdx.x >> 5;
    int which = blockIdx.z;
    int nb0 = blockIdx.x * 64;
    int mb  = (blockIdx.y * 4 + wid) * 16;
    const _Float16* WT = (which == 0) ? WqT : (which == 1) ? WkT : WvT;
    const float* bias  = (which == 0) ? bq  : (which == 1) ? bk  : bv;

    v8f acc[4] = {};
    a16f ar = load_a_f32_raw(X, D_MODEL, mb, 0, lid);
    v16h b[4];
#pragma unroll
    for (int t = 0; t < 4; ++t) b[t] = load_bt_f16(WT, D_MODEL, nb0 + t * 16, 0, lid);

#pragma unroll 2
    for (int k0 = 32; k0 < D_MODEL; k0 += 32) {
        a16f arn = load_a_f32_raw(X, D_MODEL, mb, k0, lid);   // next A in flight
        v16h bn[4];
#pragma unroll
        for (int t = 0; t < 4; ++t) bn[t] = load_bt_f16(WT, D_MODEL, nb0 + t * 16, k0, lid);
        v16h a = cvt_a(ar);                                    // current A: already resident
#pragma unroll
        for (int t = 0; t < 4; ++t) acc[t] = wmma_f16(a, b[t], acc[t]);
        ar = arn;
#pragma unroll
        for (int t = 0; t < 4; ++t) b[t] = bn[t];
    }
    {
        v16h a = cvt_a(ar);
#pragma unroll
        for (int t = 0; t < 4; ++t) acc[t] = wmma_f16(a, b[t], acc[t]);
    }

    float scale = (which == 0) ? 0.125f : 1.0f;   // 1/sqrt(64)
    int m0 = mb + ((lid >> 4) << 3);
#pragma unroll
    for (int t = 0; t < 4; ++t) {
        int n = nb0 + t * 16 + (lid & 15);
        int h = n >> 6, d = n & 63;
        float bsv = bias[n];
#pragma unroll
        for (int v = 0; v < 8; ++v) {
            int m = m0 + v;
            int bb = m >> 9, s = m & (S_ - 1);
            float val = (acc[t][v] + bsv) * scale;
            if (which == 2)
                Vt[(((size_t)(bb * N_HEAD + h) * DK + d) << 9) + s] = (_Float16)val;
            else {
                _Float16* O = (which == 0) ? Qh : Kh;
                O[(((size_t)(bb * N_HEAD + h) << 9) + s) * DK + d] = (_Float16)val;
            }
        }
    }
}

// ---- edge-key scores, batched over i:  ES[i][bh][j] = Qs[bh,i,:].ek[i,j,:] -
// Each wave: 16(bh) x 64(j) tile, K=64; all fragments preloaded.
__global__ void __launch_bounds__(128) k_escore(
    const _Float16* __restrict__ Qh, const _Float16* __restrict__ EKh,
    _Float16* __restrict__ ES) {
    int lid = threadIdx.x & 31, wid = threadIdx.x >> 5;
    int i   = blockIdx.z;
    int jb0 = blockIdx.x * 64;
    int mb  = (blockIdx.y * 4 + wid) * 16;                // bh tile
    const _Float16* Ab = Qh + (size_t)i * DK;             // A[m=bh][k=d], lda=S*64
    const _Float16* Bb = EKh + (size_t)i * (S_ * DK);     // BT[n=j][k=d], ldb=64

    v16h a0 = load_a_f16(Ab, S_ * DK, mb, 0,  lid);
    v16h a1 = load_a_f16(Ab, S_ * DK, mb, 32, lid);
    v16h b0[4], b1[4];
#pragma unroll
    for (int t = 0; t < 4; ++t) b0[t] = load_bt_f16(Bb, DK, jb0 + t * 16, 0,  lid);
#pragma unroll
    for (int t = 0; t < 4; ++t) b1[t] = load_bt_f16(Bb, DK, jb0 + t * 16, 32, lid);

    v8f acc[4] = {};
#pragma unroll
    for (int t = 0; t < 4; ++t) acc[t] = wmma_f16(a0, b0[t], acc[t]);
#pragma unroll
    for (int t = 0; t < 4; ++t) acc[t] = wmma_f16(a1, b1[t], acc[t]);

    int m0 = mb + ((lid >> 4) << 3);
    _Float16* out = ES + (size_t)i * (BH * S_);
#pragma unroll
    for (int t = 0; t < 4; ++t) {
        int n = jb0 + t * 16 + (lid & 15);
#pragma unroll
        for (int v = 0; v < 8; ++v) out[(size_t)(m0 + v) * S_ + n] = (_Float16)acc[t][v];
    }
}

// ---- fused attention per (bh, 16-row i tile) ------------------------------
__global__ void __launch_bounds__(128) k_attn(
    const _Float16* __restrict__ Qh, const _Float16* __restrict__ Kh,
    const _Float16* __restrict__ Vt, const _Float16* __restrict__ ES,
    const float* __restrict__ bias, _Float16* __restrict__ Wg,
    float* __restrict__ Ctx) {
    __shared__ __align__(32) float    sS[16 * S_];   // 32 KB score/exp rows
    __shared__ __align__(32) _Float16 sW[16 * S_];   // 16 KB f16 weights
    __shared__ float red[16][8];
    __shared__ float rmax[16], rinv[16];

    int lid = threadIdx.x & 31, wid = threadIdx.x >> 5;
    int bh = blockIdx.x;
    int ib = blockIdx.y * 16;

    const _Float16* Aq = Qh + (size_t)bh * (S_ * DK) + (size_t)ib * DK;  // lda=64
    const _Float16* Bk = Kh + (size_t)bh * (S_ * DK);                    // BT[n=j], ldb=64

    // phase 1: scores = (scaled q).kT + ES + bias.
    // A fragments loop-invariant; B pair double-buffered across j-tiles.
    {
        v16h a0 = load_a_f16(Aq, DK, 0, 0,  lid);
        v16h a1 = load_a_f16(Aq, DK, 0, 32, lid);
        v16h b0 = load_bt_f16(Bk, DK, wid * 16, 0,  lid);
        v16h b1 = load_bt_f16(Bk, DK, wid * 16, 32, lid);
        for (int jt = wid; jt < S_ / 16; jt += 4) {
            int jn = ((jt + 4) & (S_ / 16 - 1)) * 16;   // wrap: harmless redundant load
            v16h c0 = load_bt_f16(Bk, DK, jn, 0,  lid);
            v16h c1 = load_bt_f16(Bk, DK, jn, 32, lid);
            v8f acc = {};
            acc = wmma_f16(a0, b0, acc);
            acc = wmma_f16(a1, b1, acc);
            int jb = jt * 16;
            int n  = jb + (lid & 15);
            int m0 = (lid >> 4) << 3;
#pragma unroll
            for (int v = 0; v < 8; ++v) {
                int m = m0 + v;
                float sc = acc[v]
                         + (float)ES[(size_t)(ib + m) * (BH * S_) + (size_t)bh * S_ + n]
                         + bias[((size_t)bh * S_ + (ib + m)) * S_ + n];
                sS[m * S_ + n] = sc;
            }
            b0 = c0;
            b1 = c1;
        }
    }
    __syncthreads();

    // phase 2: row softmax (16 rows x 8 chunks of 64)
    int r = threadIdx.x & 15, p = threadIdx.x >> 4;
    {
        float mx = -1e30f;
        for (int j = p * 64; j < p * 64 + 64; ++j) mx = fmaxf(mx, sS[r * S_ + j]);
        red[r][p] = mx;
    }
    __syncthreads();
    if (threadIdx.x < 16) {
        float mx = red[threadIdx.x][0];
#pragma unroll
        for (int q = 1; q < 8; ++q) mx = fmaxf(mx, red[threadIdx.x][q]);
        rmax[threadIdx.x] = mx;
    }
    __syncthreads();
    {
        float mx = rmax[r], sum = 0.f;
        for (int j = p * 64; j < p * 64 + 64; ++j) {
            float e = __expf(sS[r * S_ + j] - mx);
            sS[r * S_ + j] = e;
            sum += e;
        }
        red[r][p] = sum;
    }
    __syncthreads();
    if (threadIdx.x < 16) {
        float sm = 0.f;
#pragma unroll
        for (int q = 0; q < 8; ++q) sm += red[threadIdx.x][q];
        rinv[threadIdx.x] = 1.0f / sm;
    }
    __syncthreads();
    {
        float inv = rinv[r];
        _Float16* wrow = Wg + (size_t)bh * (S_ * S_) + (size_t)(ib + r) * S_;
        for (int j = p * 64; j < p * 64 + 64; ++j) {
            _Float16 w = (_Float16)(sS[r * S_ + j] * inv);
            sW[r * S_ + j] = w;   // for phase-3 WMMA A fragments
            wrow[j] = w;          // for the batched edge-value GEMM
        }
    }
    __syncthreads();

    // phase 3: context = W @ V  (wave wid owns d-tile wid; K=512 -> 16 WMMAs)
    // A from LDS (low latency); global B double-buffered.
    {
        int nb = wid * 16;
        const _Float16* Bv = Vt + (size_t)bh * (DK * S_);   // BT[n=d][k=j], ldb=512
        v8f acc = {};
        v16h b = load_bt_f16(Bv, S_, nb, 0, lid);
#pragma unroll 4
        for (int k0 = 0; k0 < S_ - 32; k0 += 32) {
            v16h bn = load_bt_f16(Bv, S_, nb, k0 + 32, lid);
            acc = wmma_f16(load_a_f16(sW, S_, 0, k0, lid), b, acc);
            b = bn;
        }
        acc = wmma_f16(load_a_f16(sW, S_, 0, S_ - 32, lid), b, acc);

        int bb = bh >> 4, h = bh & 15;
        int d  = nb + (lid & 15);
        int m0 = (lid >> 4) << 3;
#pragma unroll
        for (int v = 0; v < 8; ++v) {
            int irow = ib + m0 + v;
            Ctx[((size_t)(bb * S_ + irow) << 10) + h * DK + d] = acc[v];
        }
    }
}

// ---- edge-value context, batched over i:  Ctx += W[bh,i,:] @ evT[i] -------
// Each wave: 16(bh) x 64(d) tile, K=512; double-buffered k-loop.
__global__ void __launch_bounds__(128) k_ectx(
    const _Float16* __restrict__ Wg, const _Float16* __restrict__ EVt,
    float* __restrict__ Ctx) {
    int lid = threadIdx.x & 31, wid = threadIdx.x >> 5;
    int i  = blockIdx.y;
    int mb = (blockIdx.x * 4 + wid) * 16;                  // bh tile
    const _Float16* Ab = Wg + (size_t)i * S_;              // A[m=bh][k=j], lda=S*S
    const _Float16* Bb = EVt + (size_t)i * (DK * S_);      // BT[n=d][k=j], ldb=512

    v8f acc[4] = {};
    v16h a = load_a_f16(Ab, S_ * S_, mb, 0, lid);
    v16h b[4];
#pragma unroll
    for (int t = 0; t < 4; ++t) b[t] = load_bt_f16(Bb, S_, t * 16, 0, lid);

#pragma unroll 2
    for (int k0 = 32; k0 < S_; k0 += 32) {
        v16h an = load_a_f16(Ab, S_ * S_, mb, k0, lid);
        v16h bn[4];
#pragma unroll
        for (int t = 0; t < 4; ++t) bn[t] = load_bt_f16(Bb, S_, t * 16, k0, lid);
#pragma unroll
        for (int t = 0; t < 4; ++t) acc[t] = wmma_f16(a, b[t], acc[t]);
        a = an;
#pragma unroll
        for (int t = 0; t < 4; ++t) b[t] = bn[t];
    }
#pragma unroll
    for (int t = 0; t < 4; ++t) acc[t] = wmma_f16(a, b[t], acc[t]);

    int m0 = mb + ((lid >> 4) << 3);
#pragma unroll
    for (int t = 0; t < 4; ++t) {
        int d = t * 16 + (lid & 15);
#pragma unroll
        for (int v = 0; v < 8; ++v) {
            int bh = m0 + v;
            int bb = bh >> 4, h = bh & 15;
            size_t idx = ((size_t)(bb * S_ + i) << 10) + h * DK + d;
            Ctx[idx] += acc[t][v];   // ordered after k_attn by stream semantics
        }
    }
}

// ---- output projection: out = Ctx[4096,1024] @ Wo + bo --------------------
// Each wave: 16(m) x 64(n) output tile, K=1024, double-buffered k-loop.
__global__ void __launch_bounds__(128) k_outproj(
    const float* __restrict__ Ctx, const _Float16* __restrict__ WoT,
    const float* __restrict__ bo, float* __restrict__ Out) {
    int lid = threadIdx.x & 31, wid = threadIdx.x >> 5;
    int nb0 = blockIdx.x * 64;
    int mb  = (blockIdx.y * 4 + wid) * 16;

    v8f acc[4] = {};
    a16f ar = load_a_f32_raw(Ctx, D_MODEL, mb, 0, lid);
    v16h b[4];
#pragma unroll
    for (int t = 0; t < 4; ++t) b[t] = load_bt_f16(WoT, D_MODEL, nb0 + t * 16, 0, lid);

#pragma unroll 2
    for (int k0 = 32; k0 < D_MODEL; k0 += 32) {
        a16f arn = load_a_f32_raw(Ctx, D_MODEL, mb, k0, lid);
        v16h bn[4];
#pragma unroll
        for (int t = 0; t < 4; ++t) bn[t] = load_bt_f16(WoT, D_MODEL, nb0 + t * 16, k0, lid);
        v16h a = cvt_a(ar);
#pragma unroll
        for (int t = 0; t < 4; ++t) acc[t] = wmma_f16(a, b[t], acc[t]);
        ar = arn;
#pragma unroll
        for (int t = 0; t < 4; ++t) b[t] = bn[t];
    }
    {
        v16h a = cvt_a(ar);
#pragma unroll
        for (int t = 0; t < 4; ++t) acc[t] = wmma_f16(a, b[t], acc[t]);
    }

    int m0 = mb + ((lid >> 4) << 3);
#pragma unroll
    for (int t = 0; t < 4; ++t) {
        int n = nb0 + t * 16 + (lid & 15);
        float bv = bo[n];
#pragma unroll
        for (int v = 0; v < 8; ++v)
            Out[(size_t)(m0 + v) * D_MODEL + n] = acc[t][v] + bv;
    }
}

// ---------------------------------------------------------------------------

extern "C" void kernel_launch(void* const* d_in, const int* in_sizes, int n_in,
                              void* d_out, int out_size, void* d_ws, size_t ws_size,
                              hipStream_t stream) {
    const float* queries     = (const float*)d_in[0];
    const float* edges_key   = (const float*)d_in[1];
    const float* edges_value = (const float*)d_in[2];
    const float* attn_bias   = (const float*)d_in[3];
    const float* Wq = (const float*)d_in[4];
    const float* bq = (const float*)d_in[5];
    const float* Wk = (const float*)d_in[6];
    const float* bk = (const float*)d_in[7];
    const float* Wv = (const float*)d_in[8];
    const float* bv = (const float*)d_in[9];
    const float* Wo = (const float*)d_in[10];
    const float* bo = (const float*)d_in[11];
    float* out = (float*)d_out;

    // bump allocator over d_ws (total ~240 MB)
    char* p = (char*)d_ws;
    auto alloc = [&](size_t bytes) {
        char* r = p;
        p += (bytes + 255) & ~(size_t)255;
        return (void*)r;
    };
    _Float16* WqT = (_Float16*)alloc((size_t)D_MODEL * D_MODEL * 2);
    _Float16* WkT = (_Float16*)alloc((size_t)D_MODEL * D_MODEL * 2);
    _Float16* WvT = (_Float16*)alloc((size_t)D_MODEL * D_MODEL * 2);
    _Float16* WoT = (_Float16*)alloc((size_t)D_MODEL * D_MODEL * 2);
    _Float16* Qh  = (_Float16*)alloc((size_t)BH * S_ * DK * 2);
    _Float16* Kh  = (_Float16*)alloc((size_t)BH * S_ * DK * 2);
    _Float16* Vt  = (_Float16*)alloc((size_t)BH * DK * S_ * 2);
    _Float16* EKh = (_Float16*)alloc((size_t)S_ * S_ * DK * 2);
    _Float16* EVt = (_Float16*)alloc((size_t)S_ * DK * S_ * 2);
    _Float16* ES  = (_Float16*)alloc((size_t)S_ * BH * S_ * 2);
    _Float16* Wg  = (_Float16*)alloc((size_t)BH * S_ * S_ * 2);
    float*    Ctx = (float*)   alloc((size_t)B_ * S_ * D_MODEL * 4);

    dim3 t16(16, 16);
    // weight transposes (f32 -> f16, [k][n] -> [n][k])
    k_transpose_cvt<<<dim3(64, 64, 1), t16, 0, stream>>>(Wq, WqT, D_MODEL, D_MODEL);
    k_transpose_cvt<<<dim3(64, 64, 1), t16, 0, stream>>>(Wk, WkT, D_MODEL, D_MODEL);
    k_transpose_cvt<<<dim3(64, 64, 1), t16, 0, stream>>>(Wv, WvT, D_MODEL, D_MODEL);
    k_transpose_cvt<<<dim3(64, 64, 1), t16, 0, stream>>>(Wo, WoT, D_MODEL, D_MODEL);
    // edges_value [i][j][d] -> EVt [i][d][j]  (batched transpose + convert)
    k_transpose_cvt<<<dim3(DK / 16, S_ / 16, S_), t16, 0, stream>>>(edges_value, EVt, S_, DK);
    // edges_key: convert only (layout is already B-fragment friendly)
    k_cvt_f16<<<8192, 256, 0, stream>>>(edges_key, EKh, (long)S_ * S_ * DK);

    // QKV projections (z: 0=Q scaled, 1=K, 2=V-transposed)
    k_qkv<<<dim3(D_MODEL / 64, 4096 / 64, 3), 128, 0, stream>>>(
        queries, WqT, WkT, WvT, bq, bk, bv, Qh, Kh, Vt);

    // edge-key scores batched over i
    k_escore<<<dim3(S_ / 64, BH / 64, S_), 128, 0, stream>>>(Qh, EKh, ES);

    // fused attention (scores + softmax + W@V)
    k_attn<<<dim3(BH, S_ / 16), 128, 0, stream>>>(Qh, Kh, Vt, ES, attn_bias, Wg, Ctx);

    // edge-value context accumulation batched over i
    k_ectx<<<dim3(BH / 64, S_), 128, 0, stream>>>(Wg, EVt, Ctx);

    // output projection
    k_outproj<<<dim3(D_MODEL / 64, 4096 / 64), 128, 0, stream>>>(Ctx, WoT, bo, out);
}
